// InvarTG_64458869178470
// MI455X (gfx1250) — matrix operand order repaired and checked by reading, not compiled, not run on hardware
//
#include <hip/hip_runtime.h>

// ---------------------------------------------------------------------------
// Problem constants (match reference)
// ---------------------------------------------------------------------------
constexpr int NN   = 10000;   // nodes
constexpr int HIDC = 256;     // hidden / in_dim
constexpr int MM   = 8;       // mixer tokens
constexpr int TT   = 3;       // timesteps
constexpr int EE   = 160000;  // edges
constexpr int AA   = 2000;    // masked nodes per step
constexpr int OUTC = 10;      // classifier outputs

// ---------------------------------------------------------------------------
// WMMA types (gfx1250, wave32)
// ---------------------------------------------------------------------------
typedef __attribute__((ext_vector_type(16))) __bf16 v16bf;
typedef __attribute__((ext_vector_type(8)))  float  v8f;

union BF16Frag {
    v16bf v;
    unsigned short u[16];
};

__device__ __forceinline__ unsigned short f2bf(float f) {
    union { float f; unsigned u; } c; c.f = f;
    unsigned u = c.u + 0x7FFFu + ((c.u >> 16) & 1u);   // round-to-nearest-even
    return (unsigned short)(u >> 16);
}

__device__ __forceinline__ float wred(float v) {
    // wave32 full reduction
    for (int o = 16; o > 0; o >>= 1) v += __shfl_xor(v, o, 32);
    return v;
}

__device__ __forceinline__ float gelu_exact(float x) {
    return 0.5f * x * (1.0f + erff(x * 0.70710678118654752f));
}

__device__ __forceinline__ float4 relu4(float4 a) {
    a.x = fmaxf(a.x, 0.f); a.y = fmaxf(a.y, 0.f);
    a.z = fmaxf(a.z, 0.f); a.w = fmaxf(a.w, 0.f);
    return a;
}

// ---------------------------------------------------------------------------
// Pack a [256,256] f32 row-major weight into a contiguous-k 16-bit B-fragment
// lane layout: PW[((tn*8 + kc)*32 + lane)*16 + e]
//   lane: col n = tn*16 + (lane&15);  g = lane>>4
//   B (32x16, 16-bit) as packed here (contiguous k, not AMD table order): lanes 0-15 hold K=0..15,
//                      lanes 16-31 hold K=16..31      ->  K = kc*32 + e + 16*g
// ---------------------------------------------------------------------------
__global__ __launch_bounds__(256) void pack_weights(const float* __restrict__ W,
                                                    unsigned short* __restrict__ PW) {
    int t    = blockIdx.x * 256 + threadIdx.x;   // 65536 total
    int e    = t & 15;
    int lane = (t >> 4) & 31;
    int kc   = (t >> 9) & 7;
    int tn   = t >> 12;
    int g    = lane >> 4;
    int K    = kc * 32 + e + 16 * g;
    int n    = tn * 16 + (lane & 15);
    PW[t] = f2bf(W[K * HIDC + n]);
}

// ---------------------------------------------------------------------------
// D[rows,256] = actOut( actIn(A[rows,256]) @ W[256,256] + bias )
// One wave computes a 16x64 strip: 1 A-frag x 4 B-frags per K-chunk,
// 8 chunks of K=32 -> 32 v_wmma_f32_16x16x32_bf16 per wave.
// Software-pipelined: chunk kc+1 loads (A raw f32 + 4 B frags) are issued
// before the 4 WMMAs of chunk kc, so VMEM latency overlaps matrix math.
// actIn: 0=none 1=relu ; actOut: 0=none 1=relu 2=gelu 3=tanh
// ---------------------------------------------------------------------------
__global__ __launch_bounds__(256) void gemm256_bf16(
    const float* __restrict__ A, const unsigned short* __restrict__ PW,
    const float* __restrict__ bias, float* __restrict__ D,
    int rowTiles, int reluIn, int actOut)
{
    int w = blockIdx.x * 8 + (threadIdx.x >> 5);
    if (w >= rowTiles * 4) return;
    int lane = threadIdx.x & 31;
    int rt = w >> 2, q = w & 3;
    int l16 = lane & 15, g = lane >> 4;
    int row = rt * 16 + l16;

    v8f acc[4];
#pragma unroll
    for (int cc = 0; cc < 4; ++cc) {
        int n = q * 64 + cc * 16 + l16;
        float bv = bias ? bias[n] : 0.0f;
        acc[cc] = (v8f){bv, bv, bv, bv, bv, bv, bv, bv};
    }

    const float* Arow = A + (size_t)row * HIDC;
    __builtin_prefetch(Arow, 0, 3);   // global_prefetch: stream this wave's A row

    // base of packed B for tn = q*4, kc = 0, this lane
    const unsigned short* PWq = PW + ((size_t)(q * 4) * 8 * 32 + lane) * 16;

    float4 na0, na1, na2, na3;        // next-chunk raw A
    v16bf  nb[4];                     // next-chunk B fragments

    auto loadA = [&](int kc) {
        const float4* p0 = reinterpret_cast<const float4*>(Arow + kc * 32 + g * 16);
        na0 = p0[0]; na1 = p0[1]; na2 = p0[2]; na3 = p0[3];
    };
    auto loadB = [&](int kc) {
#pragma unroll
        for (int cc = 0; cc < 4; ++cc)
            nb[cc] = *reinterpret_cast<const v16bf*>(PWq + cc * 4096 + kc * 512);
    };

    loadA(0);
    loadB(0);

#pragma unroll
    for (int kc = 0; kc < 8; ++kc) {
        // capture current chunk
        float4 c0 = na0, c1 = na1, c2 = na2, c3 = na3;
        v16bf  bc[4];
#pragma unroll
        for (int cc = 0; cc < 4; ++cc) bc[cc] = nb[cc];

        // issue next chunk's loads before consuming the current one
        if (kc < 7) { loadA(kc + 1); loadB(kc + 1); }

        if (reluIn) { c0 = relu4(c0); c1 = relu4(c1); c2 = relu4(c2); c3 = relu4(c3); }

        BF16Frag af;
        af.u[0]  = f2bf(c0.x); af.u[1]  = f2bf(c0.y); af.u[2]  = f2bf(c0.z); af.u[3]  = f2bf(c0.w);
        af.u[4]  = f2bf(c1.x); af.u[5]  = f2bf(c1.y); af.u[6]  = f2bf(c1.z); af.u[7]  = f2bf(c1.w);
        af.u[8]  = f2bf(c2.x); af.u[9]  = f2bf(c2.y); af.u[10] = f2bf(c2.z); af.u[11] = f2bf(c2.w);
        af.u[12] = f2bf(c3.x); af.u[13] = f2bf(c3.y); af.u[14] = f2bf(c3.z); af.u[15] = f2bf(c3.w);

#pragma unroll
        for (int cc = 0; cc < 4; ++cc) {
            acc[cc] = __builtin_amdgcn_wmma_f32_16x16x32_bf16(
                false, af.v, false, bc[cc], (short)0, acc[cc], false, false);
        }
    }

#pragma unroll
    for (int cc = 0; cc < 4; ++cc) {
        int n = q * 64 + cc * 16 + l16;
#pragma unroll
        for (int r = 0; r < 8; ++r) {
            float v = acc[cc][r];
            if      (actOut == 1) v = fmaxf(v, 0.0f);
            else if (actOut == 2) v = gelu_exact(v);
            else if (actOut == 3) v = tanhf(v);
            D[(size_t)(rt * 16 + g * 8 + r) * HIDC + n] = v;
        }
    }
}

// ---------------------------------------------------------------------------
// Mixer front half: LN1 + 8x8 token MLP (+residual) + LN2.
// One wave per node; lane owns h = j*32+lane for j=0..7.
// x_m is the circular window [N, 8, 256]; logical m -> physical (p+m)&7.
// Writes xbuf = x_m + tokenMLP (logical order), zbuf = LN2(xbuf).
// ---------------------------------------------------------------------------
__global__ __launch_bounds__(256) void mixer_kernel(
    const float* __restrict__ x_m, int p,
    const float* __restrict__ ln1g, const float* __restrict__ ln1b,
    const float* __restrict__ tw1, const float* __restrict__ tb1,
    const float* __restrict__ tw2, const float* __restrict__ tb2,
    const float* __restrict__ ln2g, const float* __restrict__ ln2b,
    float* __restrict__ xbuf, float* __restrict__ zbuf)
{
    __shared__ float sW1[64], sW2[64], sB1[8], sB2[8];
    int t = threadIdx.x;
    if (t < 64) { sW1[t] = tw1[t]; sW2[t] = tw2[t]; }
    if (t < 8)  { sB1[t] = tb1[t]; sB2[t] = tb2[t]; }
    __syncthreads();

    int node = blockIdx.x * 8 + (t >> 5);
    int lane = t & 31;

    float xv[8][8];       // [logical m][j]
    float mu[8], rs[8];

#pragma unroll
    for (int m = 0; m < 8; ++m) {
        int phys = (p + m) & 7;
        const float* src = x_m + ((size_t)node * MM + phys) * HIDC;
        float s = 0.f, s2 = 0.f;
#pragma unroll
        for (int j = 0; j < 8; ++j) {
            float v = src[j * 32 + lane];
            xv[m][j] = v; s += v; s2 += v * v;
        }
        s = wred(s); s2 = wred(s2);
        float mm = s * (1.0f / 256.0f);
        mu[m] = mm;
        rs[m] = rsqrtf(fmaxf(s2 * (1.0f / 256.0f) - mm * mm, 0.0f) + 1e-5f);
    }

    // token MLP over m, per channel h
#pragma unroll
    for (int j = 0; j < 8; ++j) {
        int h = j * 32 + lane;
        float gg = ln1g[h], bb = ln1b[h];
        float yl[8], t1[8];
#pragma unroll
        for (int m = 0; m < 8; ++m) yl[m] = (xv[m][j] - mu[m]) * rs[m] * gg + bb;
#pragma unroll
        for (int k = 0; k < 8; ++k) {
            float a = sB1[k];
#pragma unroll
            for (int m = 0; m < 8; ++m) a += yl[m] * sW1[m * 8 + k];
            t1[k] = gelu_exact(a);
        }
#pragma unroll
        for (int k = 0; k < 8; ++k) {
            float a = sB2[k];
#pragma unroll
            for (int m = 0; m < 8; ++m) a += t1[m] * sW2[m * 8 + k];
            float xb = xv[k][j] + a;           // residual
            xv[k][j] = xb;
            xbuf[((size_t)node * MM + k) * HIDC + h] = xb;
        }
    }

    // LN2 per logical token k
#pragma unroll
    for (int k = 0; k < 8; ++k) {
        float s = 0.f, s2 = 0.f;
#pragma unroll
        for (int j = 0; j < 8; ++j) { float v = xv[k][j]; s += v; s2 += v * v; }
        s = wred(s); s2 = wred(s2);
        float mm = s * (1.0f / 256.0f);
        float rr = rsqrtf(fmaxf(s2 * (1.0f / 256.0f) - mm * mm, 0.0f) + 1e-5f);
#pragma unroll
        for (int j = 0; j < 8; ++j) {
            int h = j * 32 + lane;
            zbuf[((size_t)node * MM + k) * HIDC + h] = (xv[k][j] - mm) * rr * ln2g[h] + ln2b[h];
        }
    }
}

// v_out[n,h] = mean_k( xbuf[n,k,h] + g2[n,k,h] )
__global__ __launch_bounds__(256) void mean_kernel(
    const float* __restrict__ xbuf, const float* __restrict__ g2,
    float* __restrict__ vout)
{
    int i = blockIdx.x * 256 + threadIdx.x;   // N*256
    int n = i >> 8, h = i & 255;
    float s = 0.f;
#pragma unroll
    for (int k = 0; k < 8; ++k) {
        size_t idx = ((size_t)n * MM + k) * HIDC + h;
        s += xbuf[idx] + g2[idx];
    }
    vout[i] = s * 0.125f;
}

// ---------------------------------------------------------------------------
// GCN helpers
// ---------------------------------------------------------------------------
__global__ __launch_bounds__(256) void deg_kernel(const int* __restrict__ dst,
                                                  float* __restrict__ deg, int ne) {
    int e = blockIdx.x * 256 + threadIdx.x;
    if (e < ne) atomicAdd(&deg[dst[e]], 1.0f);
}

__global__ __launch_bounds__(256) void dinv_kernel(const float* __restrict__ deg,
                                                   float* __restrict__ dinv, int n) {
    int i = blockIdx.x * 256 + threadIdx.x;
    if (i < n) dinv[i] = rsqrtf(deg[i] + 1.0f);   // +1 self loop, always >= 1
}

// out[i,:] = xw[i,:] * dinv[i]^2 + b  (self-loop term + bias); rowStride in floats
__global__ __launch_bounds__(256) void gcn_init(const float* __restrict__ xw,
                                                const float* __restrict__ dinv,
                                                const float* __restrict__ b,
                                                float* __restrict__ out, int rowStride) {
    int i = blockIdx.x, h = threadIdx.x;
    float di = dinv[i];
    out[(size_t)i * rowStride + h] = xw[(size_t)i * HIDC + h] * di * di + b[h];
}

// out[dst,:] += xw[src,:] * dinv[src]*dinv[dst]
__global__ __launch_bounds__(256) void gcn_edge(const int* __restrict__ src,
                                                const int* __restrict__ dst,
                                                const float* __restrict__ dinv,
                                                const float* __restrict__ xw,
                                                float* __restrict__ out, int rowStride) {
    int e = blockIdx.x, h = threadIdx.x;
    int s = src[e], d = dst[e];
    float norm = dinv[s] * dinv[d];
    atomicAdd(&out[(size_t)d * rowStride + h], xw[(size_t)s * HIDC + h] * norm);
}

__global__ __launch_bounds__(256) void zero_rows(const int* __restrict__ nid,
                                                 float* __restrict__ out, int rowStride) {
    int a = blockIdx.x, h = threadIdx.x;
    out[(size_t)nid[a] * rowStride + h] = 0.0f;
}

// y_hat[r,o] = relu( c1[r,:] . w[:,o] + b[o] )  (256 -> 10 head)
__global__ __launch_bounds__(256) void cls2_kernel(const float* __restrict__ c1,
                                                   const float* __restrict__ w,
                                                   const float* __restrict__ b,
                                                   float* __restrict__ out, int rows) {
    int id = blockIdx.x * 256 + threadIdx.x;
    if (id >= rows * OUTC) return;
    int r = id / OUTC, o = id - r * OUTC;
    const float* cr = c1 + (size_t)r * HIDC;
    float s = b[o];
#pragma unroll 8
    for (int k = 0; k < HIDC; ++k) s += cr[k] * w[k * OUTC + o];
    out[id] = fmaxf(s, 0.0f);
}

// ---------------------------------------------------------------------------
// Host orchestration
// ---------------------------------------------------------------------------
extern "C" void kernel_launch(void* const* d_in, const int* in_sizes, int n_in,
                              void* d_out, int out_size, void* d_ws, size_t ws_size,
                              hipStream_t stream) {
    (void)in_sizes; (void)n_in; (void)out_size; (void)ws_size;

    const int*   edge_t  = (const int*)d_in[0];    // [T,2,E]
    const int*   node_ids= (const int*)d_in[1];    // [T,A]
    const float* emb     = (const float*)d_in[3];
    const float* gcn_w1  = (const float*)d_in[4];
    const float* gcn_b1  = (const float*)d_in[5];
    const float* gcn_w2  = (const float*)d_in[6];
    const float* gcn_b2  = (const float*)d_in[7];
    const float* ln1_g   = (const float*)d_in[8];
    const float* ln1_b   = (const float*)d_in[9];
    const float* tok_w1  = (const float*)d_in[10];
    const float* tok_b1  = (const float*)d_in[11];
    const float* tok_w2  = (const float*)d_in[12];
    const float* tok_b2  = (const float*)d_in[13];
    const float* ln2_g   = (const float*)d_in[14];
    const float* ln2_b   = (const float*)d_in[15];
    const float* ch_w1   = (const float*)d_in[16];
    const float* ch_b1   = (const float*)d_in[17];
    const float* ch_w2   = (const float*)d_in[18];
    const float* ch_b2   = (const float*)d_in[19];
    const float* pred_w  = (const float*)d_in[20];
    const float* pred_b  = (const float*)d_in[21];
    const float* cls1_w  = (const float*)d_in[22];
    const float* cls1_b  = (const float*)d_in[23];
    const float* cls2_w  = (const float*)d_in[24];
    const float* cls2_b  = (const float*)d_in[25];

    char* base = (char*)d_ws;
    size_t off = 0;
    auto alloc = [&](size_t bytes) -> char* {
        char* p = base + off;
        off = (off + bytes + 255) & ~(size_t)255;
        return p;
    };

    constexpr size_t PW_ELEMS = 16 * 8 * 32 * 16;                 // 65536 bf16
    unsigned short* pw[6];
    for (int i = 0; i < 6; ++i) pw[i] = (unsigned short*)alloc(PW_ELEMS * 2);

    const size_t big  = (size_t)NN * MM * HIDC * sizeof(float);   // 81.92 MB
    const size_t midv = (size_t)TT * NN * HIDC * sizeof(float);   // 30.72 MB
    const size_t nh   = (size_t)NN * HIDC * sizeof(float);        // 10.24 MB

    float* x_m  = (float*)alloc(big);
    float* xbuf = (float*)alloc(big);
    float* zbuf = (float*)alloc(big);
    float* g1   = (float*)alloc(big);
    float* vbuf = (float*)alloc(midv);
    float* c1   = (float*)alloc(midv);
    float* xw1  = (float*)alloc(nh);
    float* out1 = (float*)alloc(nh);
    float* xw2  = (float*)alloc(nh);
    float* deg  = (float*)alloc((size_t)NN * sizeof(float));
    float* dinv = (float*)alloc((size_t)NN * sizeof(float));

    float* hout = (float*)d_out;                       // h: [T*N, 256]
    float* yout = hout + (size_t)TT * NN * HIDC;       // y_hat: [T*N, 10]

    // Pack the six 256x256 weight matrices into WMMA B-fragment order (bf16)
    const float* Ws[6] = {gcn_w1, gcn_w2, ch_w1, ch_w2, pred_w, cls1_w};
    for (int i = 0; i < 6; ++i)
        pack_weights<<<256, 256, 0, stream>>>(Ws[i], pw[i]);

    hipMemsetAsync(x_m, 0, big, stream);

    // emb @ gcn_w1 is time-invariant: hoist out of the loop
    const int RT_N  = NN / 16;                         // 625
    const int RT_NM = NN * MM / 16;                    // 5000
    const int RT_TN = TT * NN / 16;                    // 1875
    auto gblocks = [](int rt) { return (rt * 4 + 7) / 8; };

    gemm256_bf16<<<gblocks(RT_N), 256, 0, stream>>>(emb, pw[0], nullptr, xw1, RT_N, 0, 0);

    for (int t = 0; t < TT; ++t) {
        // ---- mixer forecast from current window (head p = t, before update)
        mixer_kernel<<<NN / 8, 256, 0, stream>>>(
            x_m, t, ln1_g, ln1_b, tok_w1, tok_b1, tok_w2, tok_b2,
            ln2_g, ln2_b, xbuf, zbuf);
        gemm256_bf16<<<gblocks(RT_NM), 256, 0, stream>>>(zbuf, pw[2], ch_b1, g1,   RT_NM, 0, 2); // gelu
        gemm256_bf16<<<gblocks(RT_NM), 256, 0, stream>>>(g1,   pw[3], ch_b2, zbuf, RT_NM, 0, 0); // z (g2)
        mean_kernel<<<(NN * HIDC + 255) / 256, 256, 0, stream>>>(xbuf, zbuf, vbuf + (size_t)t * NN * HIDC);

        // ---- 2-layer GCN for this step's graph
        const int* src = edge_t + (size_t)t * 2 * EE;
        const int* dst = src + EE;
        hipMemsetAsync(deg, 0, (size_t)NN * sizeof(float), stream);
        deg_kernel<<<(EE + 255) / 256, 256, 0, stream>>>(dst, deg, EE);
        dinv_kernel<<<(NN + 255) / 256, 256, 0, stream>>>(deg, dinv, NN);

        gcn_init<<<NN, 256, 0, stream>>>(xw1, dinv, gcn_b1, out1, HIDC);
        gcn_edge<<<EE, 256, 0, stream>>>(src, dst, dinv, xw1, out1, HIDC);

        gemm256_bf16<<<gblocks(RT_N), 256, 0, stream>>>(out1, pw[1], nullptr, xw2, RT_N, /*reluIn=*/1, 0);

        // write conv-2 output straight into circular window slot (= t, head p=t)
        int slot = t & 7;
        float* slotp = x_m + slot * HIDC;              // rowStride = M*HID = 2048
        gcn_init<<<NN, 256, 0, stream>>>(xw2, dinv, gcn_b2, slotp, MM * HIDC);
        gcn_edge<<<EE, 256, 0, stream>>>(src, dst, dinv, xw2, slotp, MM * HIDC);
        zero_rows<<<AA, 256, 0, stream>>>(node_ids + (size_t)t * AA, slotp, MM * HIDC);
    }

    // ---- invariant predictor over all T steps at once
    gemm256_bf16<<<gblocks(RT_TN), 256, 0, stream>>>(vbuf, pw[4], pred_b, hout, RT_TN, 0, 3); // tanh -> h
    gemm256_bf16<<<gblocks(RT_TN), 256, 0, stream>>>(hout, pw[5], cls1_b, c1,   RT_TN, 0, 0);
    cls2_kernel<<<(TT * NN * OUTC + 255) / 256, 256, 0, stream>>>(c1, cls2_w, cls2_b, yout, TT * NN);
}
